// OCAB_44461501448699
// MI455X (gfx1250) — matrix operand order repaired and checked by reading, not compiled, not run on hardware
//
#include <hip/hip_runtime.h>
#include <hip/hip_bf16.h>

#define DIMC   384
#define HEADS  6
#define HDD    64
#define HIDN   768
#define WSZ    16
#define OWSZ   24
#define IMG    192
#define NWIN   12
#define NTOK   (IMG*IMG)      // 36864
#define QKVN   (3*DIMC)       // 1152
#define TABLEN 1521

typedef __attribute__((ext_vector_type(16))) __bf16 bf16x16;
typedef __attribute__((ext_vector_type(8)))  float  f32x8;
typedef __attribute__((ext_vector_type(4)))  unsigned int u32x4;
typedef __attribute__((ext_vector_type(8)))  unsigned int u32x8;

union FragU { bf16x16 v; unsigned int u[8]; };
union V8    { uint4 q; unsigned short s[8]; };

__device__ __forceinline__ unsigned short f2bf(float x) {
    unsigned int u = __float_as_uint(x);
    u += 0x7FFFu + ((u >> 16) & 1u);
    return (unsigned short)(u >> 16);
}

__device__ __forceinline__ f32x8 zero8() {
    f32x8 z;
#pragma unroll
    for (int i = 0; i < 8; ++i) z[i] = 0.0f;
    return z;
}

// A-fragment (16x32 bf16): lane holds row M = lane&15 (ISA interleaved K map).
__device__ __forceinline__ bf16x16 load_fragA(const unsigned short* p, int g) {
    FragU f;
#pragma unroll
    for (int v = 0; v < 8; ++v) {
        int k = ((v < 4) ? (2 * v) : (2 * v + 8)) + 8 * g;
        f.u[v] = *reinterpret_cast<const unsigned int*>(p + k);
    }
    return f.v;
}

// B-fragment (32x16 bf16): lane holds col N = lane&15 (ISA interleaved K map).
__device__ __forceinline__ bf16x16 load_fragB(const unsigned short* p, int g) {
    FragU f;
#pragma unroll
    for (int v = 0; v < 8; ++v) {
        int k = ((v < 4) ? (2 * v) : (2 * v + 8)) + 8 * g;
        f.u[v] = *reinterpret_cast<const unsigned int*>(p + k);
    }
    return f.v;
}

#define WMMA_BF16(a, b, c) \
    __builtin_amdgcn_wmma_f32_16x16x32_bf16(false, (a), false, (b), (short)0, (c), false, false)

// ---------------------------------------------------------------------------
// TDM: DMA a 64-row x 32-elem bf16 tile (row stride K elems) from global
// into LDS (rows packed contiguously, 64 B apiece). D# built per ISA ch.8:
// group0 = {count|flags, lds_addr, global_addr[56:0] | type=2<<30},
// group1 = {mask/data_size, dims/tiles/strides}. 2-group form (<=2D tensor).
// ---------------------------------------------------------------------------
__device__ __forceinline__ void tdm_load_b_tile(const unsigned short* lds_dst,
                                                const unsigned short* gsrc,
                                                int K) {
    unsigned long long ga = (unsigned long long)(uintptr_t)gsrc;
    unsigned lds = (unsigned)(uintptr_t)lds_dst;   // LDS aperture: addr[31:0]
    u32x4 g0;
    g0[0] = 1u;                                            // count=1, no gather
    g0[1] = lds;                                           // lds_addr (bytes)
    g0[2] = (unsigned)(ga & 0xFFFFFFFFu);                  // global_addr lo
    g0[3] = ((unsigned)(ga >> 32) & 0x01FFFFFFu) | (2u << 30); // addr hi | type=2
    u32x8 g1;
    g1[0] = 0x00010000u;          // workgroup_mask=0, data_size=1 (2 bytes)
    g1[1] = 0u;                   // abar addr=0, tensor_dim0[15:0]=0
    g1[2] = 0x4000u;              // tensor_dim0 = 1<<30 (huge, no OOB), td1 lo=0
    g1[3] = 0x4000u | (32u << 16);// tensor_dim1 = 1<<30, tile_dim0 = 32 elems
    g1[4] = 64u;                  // tile_dim1 = 64, tile_dim2 = 0
    g1[5] = (unsigned)K;          // tensor_dim0_stride lo (elems)
    g1[6] = 0u;                   // stride0 hi, stride1 lo
    g1[7] = 0u;                   // stride1 hi
    asm volatile("tensor_load_to_lds %0, %1" :: "s"(g0), "s"(g1) : "memory");
}

// ---------------------------------------------------------------------------
// Weight convert + transpose: w[K][N] f32 -> wt[N][K] bf16 (N-major)
// ---------------------------------------------------------------------------
__global__ void wconv_t(const float* __restrict__ w, unsigned short* __restrict__ wt,
                        int K, int N) {
    int i = blockIdx.x * blockDim.x + threadIdx.x;
    if (i >= K * N) return;
    int k = i / N, n = i - k * N;
    wt[(size_t)n * K + k] = f2bf(w[i]);
}

// ---------------------------------------------------------------------------
// LayerNorm over 384 columns -> bf16. One row per block of 128 threads.
// ---------------------------------------------------------------------------
__global__ void ln_to_bf16(const float* __restrict__ x, const float* __restrict__ gma,
                           const float* __restrict__ bta, unsigned short* __restrict__ out) {
    int row = blockIdx.x;
    int tid = threadIdx.x;                // 128 threads
    const float* xr = x + (size_t)row * DIMC;
    float v0 = xr[tid], v1 = xr[tid + 128], v2 = xr[tid + 256];
    float s = v0 + v1 + v2;
    __shared__ float red[8];
#pragma unroll
    for (int m = 16; m >= 1; m >>= 1) s += __shfl_xor(s, m, 32);
    if ((tid & 31) == 0) red[tid >> 5] = s;
    __syncthreads();
    float mean = (red[0] + red[1] + red[2] + red[3]) * (1.0f / DIMC);
    float d0 = v0 - mean, d1 = v1 - mean, d2 = v2 - mean;
    float q = d0 * d0 + d1 * d1 + d2 * d2;
#pragma unroll
    for (int m = 16; m >= 1; m >>= 1) q += __shfl_xor(q, m, 32);
    if ((tid & 31) == 0) red[4 + (tid >> 5)] = q;
    __syncthreads();
    float var = (red[4] + red[5] + red[6] + red[7]) * (1.0f / DIMC);
    float inv = rsqrtf(var + 1e-5f);
    unsigned short* orow = out + (size_t)row * DIMC;
    orow[tid]       = f2bf(d0 * inv * gma[tid]       + bta[tid]);
    orow[tid + 128] = f2bf(d1 * inv * gma[tid + 128] + bta[tid + 128]);
    orow[tid + 256] = f2bf(d2 * inv * gma[tid + 256] + bta[tid + 256]);
}

// ---------------------------------------------------------------------------
// Generic WMMA GEMM: C[M,N] = A[M,K](bf16) * B[K,N] (Bt stored [N,K] bf16)
// block = 256 threads = 8 waves; tile = 128 rows x 64 cols; wave = 16x64.
// B tiles are DMA'd by the Tensor Data Mover, double-buffered:
// TDM(tile i+1) issued before computing tile i; s_wait_tensorcnt + barrier.
// mode 0: out_bf16 = acc + bias
// mode 1: out_bf16 = gelu(acc + bias)
// mode 2: out_f32  = acc + bias + res
// ---------------------------------------------------------------------------
__global__ void __launch_bounds__(256)
gemm_wmma(const unsigned short* __restrict__ A,
          const unsigned short* __restrict__ Bt,
          const float* __restrict__ bias,
          const float* __restrict__ res,
          float* __restrict__ outf,
          unsigned short* __restrict__ outb,
          int M, int N, int K, int mode) {
    __shared__ unsigned short bs[2][64 * 32];    // double-buffered B tile, 2x4 KB
    int wave = threadIdx.x >> 5;
    int lane = threadIdx.x & 31;
    int g = lane >> 4, l15 = lane & 15;
    int rowbase = blockIdx.x * 128 + wave * 16;
    int colbase = blockIdx.y * 64;

    f32x8 acc[4];
#pragma unroll
    for (int t = 0; t < 4; ++t) acc[t] = zero8();

    const unsigned short* arow = A + (size_t)(rowbase + l15) * K;
    const unsigned short* btile = Bt + (size_t)colbase * K;
    int nk = K / 32;

    if (threadIdx.x < 32) tdm_load_b_tile(&bs[0][0], btile, K);

    for (int it = 0; it < nk; ++it) {
        int cur = it & 1;
        if (it + 1 < nk) {
            if (threadIdx.x < 32)
                tdm_load_b_tile(&bs[cur ^ 1][0], btile + (it + 1) * 32, K);
            __builtin_amdgcn_s_wait_tensorcnt(1);   // tile `it` landed
        } else {
            __builtin_amdgcn_s_wait_tensorcnt(0);
        }
        __syncthreads();
        int k0 = it * 32;
        if (k0 + 32 < K) __builtin_prefetch(arow + k0 + 32, 0, 1);
        bf16x16 afrag = load_fragA(arow + k0, g);
#pragma unroll
        for (int t = 0; t < 4; ++t) {
            bf16x16 bfrag = load_fragB(&bs[cur][(t * 16 + l15) * 32], g);
            acc[t] = WMMA_BF16(afrag, bfrag, acc[t]);
        }
        __syncthreads();    // all reads of bs[cur] done before it becomes a TDM target
    }

#pragma unroll
    for (int t = 0; t < 4; ++t) {
#pragma unroll
        for (int r = 0; r < 8; ++r) {
            int row = rowbase + r + 8 * g;
            int col = colbase + t * 16 + l15;
            float v = acc[t][r] + bias[col];
            size_t o = (size_t)row * N + col;
            if (mode == 1) v = 0.5f * v * (1.0f + erff(v * 0.70710678118f));
            if (mode == 2) outf[o] = v + res[o];
            else           outb[o] = f2bf(v);
        }
    }
}

// ---------------------------------------------------------------------------
// Overlapping-window attention, flash-style online softmax.
// grid = (144 windows, 6 heads, 2 query halves); block = 256 threads = 8 waves.
// ---------------------------------------------------------------------------
__global__ void __launch_bounds__(256)
attn_kernel(const unsigned short* __restrict__ qkv,
            const float* __restrict__ rpb,             // [1521][6]
            unsigned short* __restrict__ outb) {
    int win  = blockIdx.x;
    int head = blockIdx.y;
    int qhalf = blockIdx.z;
    int wr = win / NWIN, wc = win - wr * NWIN;
    int wave = threadIdx.x >> 5, lane = threadIdx.x & 31;
    int g = lane >> 4, l15 = lane & 15;

    __shared__ unsigned short kbuf[64 * 64];    // [key][d]   8 KB
    __shared__ unsigned short vtbuf[64 * 64];   // [d][key]   8 KB
    __shared__ unsigned short pbuf[8][16 * 64]; // per-wave P 16 KB
    __shared__ float biasb[TABLEN];             // head's bias column ~6 KB

    for (int i = threadIdx.x; i < TABLEN; i += 256) biasb[i] = rpb[i * HEADS + head];

    int stripe = qhalf * 8 + wave;              // 0..15
    int qbase = stripe * 16;
    int q = qbase + l15;
    int qi = q >> 4, qj = q & 15;
    int qtok = (wr * WSZ + qi) * IMG + (wc * WSZ + qj);
    const unsigned short* qrow = qkv + (size_t)qtok * QKVN + head * HDD;
    bf16x16 qf0 = load_fragA(qrow, g);
    bf16x16 qf1 = load_fragA(qrow + 32, g);

    f32x8 oacc[4];
#pragma unroll
    for (int t = 0; t < 4; ++t) oacc[t] = zero8();
    float mrun[8], lrun[8];
#pragma unroll
    for (int r = 0; r < 8; ++r) { mrun[r] = -1e30f; lrun[r] = 0.0f; }

    for (int ch = 0; ch < 9; ++ch) {
        __syncthreads();
        // stage k [key][d] and v transposed [d][key]; 16-byte segments
        for (int i = threadIdx.x; i < 512; i += 256) {
            int key = i >> 3, d8 = (i & 7) << 3;
            int kk = ch * 64 + key;
            int oi = kk / OWSZ, oj = kk - oi * OWSZ;
            int gi = wr * WSZ - 4 + oi, gj = wc * WSZ - 4 + oj;
            V8 kd, vd;
            if (gi >= 0 && gi < IMG && gj >= 0 && gj < IMG) {
                size_t base = (size_t)(gi * IMG + gj) * QKVN + head * HDD + d8;
                kd.q = *(const uint4*)(qkv + base + DIMC);
                vd.q = *(const uint4*)(qkv + base + 2 * DIMC);
            } else {
                kd.q = make_uint4(0u, 0u, 0u, 0u);
                vd.q = make_uint4(0u, 0u, 0u, 0u);
            }
            *(uint4*)&kbuf[key * 64 + d8] = kd.q;
#pragma unroll
            for (int d = 0; d < 8; ++d) vtbuf[(d8 + d) * 64 + key] = vd.s[d];
        }
        __syncthreads();

        // S = q k^T  (4 key tiles x 2 k-steps over d)
        f32x8 s[4];
#pragma unroll
        for (int t = 0; t < 4; ++t) {
            s[t] = zero8();
            const unsigned short* krow = &kbuf[(t * 16 + l15) * 64];
            bf16x16 kf0 = load_fragB(krow, g);
            bf16x16 kf1 = load_fragB(krow + 32, g);
            s[t] = WMMA_BF16(qf0, kf0, s[t]);
            s[t] = WMMA_BF16(qf1, kf1, s[t]);
        }

        // scale + relative-position bias; per-row running max
        float rowmax[8];
#pragma unroll
        for (int r = 0; r < 8; ++r) rowmax[r] = -1e30f;
#pragma unroll
        for (int t = 0; t < 4; ++t) {
            int kk = ch * 64 + t * 16 + l15;
            int oi = kk / OWSZ, oj = kk - oi * OWSZ;
#pragma unroll
            for (int r = 0; r < 8; ++r) {
                int qq = qbase + r + 8 * g;
                int qqi = qq >> 4, qqj = qq & 15;
                int bidx = (qqi - oi + 23) * 39 + (qqj - oj + 23);
                float sv = s[t][r] * 0.125f + biasb[bidx];
                s[t][r] = sv;
                rowmax[r] = fmaxf(rowmax[r], sv);
            }
        }
#pragma unroll
        for (int r = 0; r < 8; ++r) {
#pragma unroll
            for (int m = 8; m >= 1; m >>= 1)
                rowmax[r] = fmaxf(rowmax[r], __shfl_xor(rowmax[r], m, 32));
        }

        // online-softmax update
        float corr[8], rs[8];
#pragma unroll
        for (int r = 0; r < 8; ++r) {
            float mn = fmaxf(mrun[r], rowmax[r]);
            corr[r] = __expf(mrun[r] - mn);
            mrun[r] = mn;
            lrun[r] *= corr[r];
            rs[r] = 0.0f;
        }
#pragma unroll
        for (int t = 0; t < 4; ++t) {
#pragma unroll
            for (int r = 0; r < 8; ++r) {
                float p = __expf(s[t][r] - mrun[r]);
                s[t][r] = p;
                rs[r] += p;
            }
        }
#pragma unroll
        for (int r = 0; r < 8; ++r) {
#pragma unroll
            for (int m = 8; m >= 1; m >>= 1) rs[r] += __shfl_xor(rs[r], m, 32);
            lrun[r] += rs[r];
        }
#pragma unroll
        for (int t = 0; t < 4; ++t)
#pragma unroll
            for (int r = 0; r < 8; ++r) oacc[t][r] *= corr[r];

        // D-layout P -> LDS -> A-layout fragments
        unsigned short* pb = pbuf[wave];
#pragma unroll
        for (int t = 0; t < 4; ++t)
#pragma unroll
            for (int r = 0; r < 8; ++r)
                pb[(r + 8 * g) * 64 + t * 16 + l15] = f2bf(s[t][r]);
        __syncthreads();

        // O += P * V
        bf16x16 pf0 = load_fragA(&pb[l15 * 64], g);
        bf16x16 pf1 = load_fragA(&pb[l15 * 64 + 32], g);
#pragma unroll
        for (int t = 0; t < 4; ++t) {
            const unsigned short* vrow = &vtbuf[(t * 16 + l15) * 64];
            bf16x16 vf0 = load_fragB(vrow, g);
            bf16x16 vf1 = load_fragB(vrow + 32, g);
            oacc[t] = WMMA_BF16(pf0, vf0, oacc[t]);
            oacc[t] = WMMA_BF16(pf1, vf1, oacc[t]);
        }
    }

    // normalize and scatter back to token layout (bf16 for proj GEMM)
#pragma unroll
    for (int t = 0; t < 4; ++t) {
#pragma unroll
        for (int r = 0; r < 8; ++r) {
            int qq = qbase + r + 8 * g;
            int qqi = qq >> 4, qqj = qq & 15;
            int tok = (wr * WSZ + qqi) * IMG + (wc * WSZ + qqj);
            int col = head * HDD + t * 16 + l15;
            outb[(size_t)tok * DIMC + col] = f2bf(oacc[t][r] / lrun[r]);
        }
    }
}

// ---------------------------------------------------------------------------
extern "C" void kernel_launch(void* const* d_in, const int* in_sizes, int n_in,
                              void* d_out, int out_size, void* d_ws, size_t ws_size,
                              hipStream_t stream) {
    (void)in_sizes; (void)n_in; (void)out_size; (void)ws_size;
    const float* x      = (const float*)d_in[0];
    const float* n1g    = (const float*)d_in[1];
    const float* n1b    = (const float*)d_in[2];
    const float* qkv_w  = (const float*)d_in[3];
    const float* qkv_b  = (const float*)d_in[4];
    const float* rpb    = (const float*)d_in[5];
    const float* proj_w = (const float*)d_in[6];
    const float* proj_b = (const float*)d_in[7];
    const float* n2g    = (const float*)d_in[8];
    const float* n2b    = (const float*)d_in[9];
    const float* fc1_w  = (const float*)d_in[10];
    const float* fc1_b  = (const float*)d_in[11];
    const float* fc2_w  = (const float*)d_in[12];
    const float* fc2_b  = (const float*)d_in[13];
    float* out = (float*)d_out;
    char* ws = (char*)d_ws;

    const size_t szR0 = (size_t)NTOK * DIMC * 2;   // xn / attn_out
    const size_t szR1 = (size_t)NTOK * QKVN * 2;   // qkv | (h + hid)
    const size_t szR2 = (size_t)NTOK * DIMC * 4;   // x2 f32
    unsigned short* xn       = (unsigned short*)(ws);
    unsigned short* attn_out = xn;                 // alias (xn dead)
    unsigned short* qkvb     = (unsigned short*)(ws + szR0);
    unsigned short* hbuf     = qkvb;               // alias (qkv dead)
    unsigned short* hid      = (unsigned short*)(ws + szR0 + (size_t)NTOK * DIMC * 2);
    float*          x2       = (float*)(ws + szR0 + szR1);
    unsigned short* wts      = (unsigned short*)(ws + szR0 + szR1 + szR2);
    unsigned short* qkvT  = wts;
    unsigned short* projT = qkvT + (size_t)DIMC * QKVN;
    unsigned short* fc1T  = projT + (size_t)DIMC * DIMC;
    unsigned short* fc2T  = fc1T  + (size_t)DIMC * HIDN;

    {
        int n;
        n = DIMC * QKVN; wconv_t<<<(n + 255) / 256, 256, 0, stream>>>(qkv_w, qkvT, DIMC, QKVN);
        n = DIMC * DIMC; wconv_t<<<(n + 255) / 256, 256, 0, stream>>>(proj_w, projT, DIMC, DIMC);
        n = DIMC * HIDN; wconv_t<<<(n + 255) / 256, 256, 0, stream>>>(fc1_w, fc1T, DIMC, HIDN);
        n = HIDN * DIMC; wconv_t<<<(n + 255) / 256, 256, 0, stream>>>(fc2_w, fc2T, HIDN, DIMC);
    }
    ln_to_bf16<<<NTOK, 128, 0, stream>>>(x, n1g, n1b, xn);
    gemm_wmma<<<dim3(NTOK / 128, QKVN / 64), 256, 0, stream>>>(
        xn, qkvT, qkv_b, nullptr, nullptr, qkvb, NTOK, QKVN, DIMC, 0);
    attn_kernel<<<dim3(NWIN * NWIN, HEADS, 2), 256, 0, stream>>>(qkvb, rpb, attn_out);
    gemm_wmma<<<dim3(NTOK / 128, DIMC / 64), 256, 0, stream>>>(
        attn_out, projT, proj_b, x, x2, nullptr, NTOK, DIMC, DIMC, 2);
    ln_to_bf16<<<NTOK, 128, 0, stream>>>(x2, n2g, n2b, hbuf);
    gemm_wmma<<<dim3(NTOK / 128, HIDN / 64), 256, 0, stream>>>(
        hbuf, fc1T, fc1_b, nullptr, nullptr, hid, NTOK, HIDN, DIMC, 1);
    gemm_wmma<<<dim3(NTOK / 128, DIMC / 64), 256, 0, stream>>>(
        hid, fc2T, fc2_b, x2, out, nullptr, NTOK, DIMC, HIDN, 2);
}
